// RNNClassifier_44822278701459
// MI455X (gfx1250) — matrix operand order repaired and checked by reading, not compiled, not run on hardware
//
#include <hip/hip_runtime.h>

// ---------------------------------------------------------------------------
// MI455X (gfx1250) bidirectional-GRU classifier.
// Strategy: precompute x-projections with big bf16 WMMA GEMMs (parallel),
// keep the serial recurrence to h@Wh only (12 WMMAs/wave/step, 32 chains,
// Wh staged in LDS), WMMA attention scoring, VALU softmax + tiny head.
// ---------------------------------------------------------------------------

typedef __attribute__((ext_vector_type(16))) __bf16 v16bf;
typedef __attribute__((ext_vector_type(8)))  float  v8f;

union Frag16 {
  v16bf v;
  unsigned int u[8];
  unsigned short s[16];
};

__device__ __forceinline__ unsigned short f2bf(float f) {
  unsigned int u = __float_as_uint(f);
  u += 0x7FFFu + ((u >> 16) & 1u);   // round-to-nearest-even
  return (unsigned short)(u >> 16);
}
__device__ __forceinline__ float bf2f(unsigned short h) {
  return __uint_as_float(((unsigned int)h) << 16);
}

// 16-bit A-matrix 16x32 K map (ISA 7.12.2): lanes 0-15: v0-3 K=0..7, v4-7 K=16..23;
// lanes 16-31: +8.  Byte offset with the 8*half term folded into the base ptr:
__device__ __forceinline__ int ka_byte(int p) {          // *2 bytes, no half term
  return (p < 4 ? 4 * p : 32 + 4 * (p - 4));
}
// 16-bit B-matrix 32x16 K map (same as A): v0-3 K=0..7, v4-7 K=16..23; lanes 16-31 hold K+8.
__device__ __forceinline__ int kb_of(int p, int half) {
  return 2 * p + (p >= 4 ? 8 : 0) + 8 * half;
}

__device__ __forceinline__ v8f wmma_bf16(v16bf a, v16bf b, v8f c) {
  return __builtin_amdgcn_wmma_f32_16x16x32_bf16(
      /*neg_a=*/false, a, /*neg_b=*/false, b,
      /*c_mod=*/(short)0, c, /*reuse_a=*/false, /*reuse_b=*/false);
}

// ---------------------------------------------------------------------------
// Pack a (fused) f32 weight matrix into WMMA B-fragment tile layout (bf16).
// Tile (kt,nt) = 32 lanes x 16 bf16 contiguous (1024B). n < ncolsA -> srcA,
// else srcB (cand part). k >= kvalid -> zero pad (E=300 padded to 320).
// ---------------------------------------------------------------------------
__global__ void pack_tiles(const float* __restrict__ srcA, const float* __restrict__ srcB,
                           int ncolsA, int ncolsB, int NT, int row_off, int kvalid,
                           unsigned short* __restrict__ out) {
  int tid = threadIdx.x;
  int lane = tid >> 4, i = tid & 15;
  int half = lane >> 4;
  int kt = blockIdx.x / NT, nt = blockIdx.x % NT;
  int k = kt * 32 + kb_of(i >> 1, half) + (i & 1);
  int n = nt * 16 + (lane & 15);
  float v = 0.f;
  if (k < kvalid) {
    if (n < ncolsA) v = srcA[(size_t)(row_off + k) * ncolsA + n];
    else            v = srcB[(size_t)(row_off + k) * ncolsB + (n - ncolsA)];
  }
  out[(size_t)blockIdx.x * 512 + lane * 16 + i] = f2bf(v);
}

// ---------------------------------------------------------------------------
// Embedding gather + f32 -> bf16, K padded 300 -> 320 with zeros.
// ---------------------------------------------------------------------------
__global__ void embed_cast(const int* __restrict__ tok, const float* __restrict__ emb,
                           unsigned short* __restrict__ out, int nrows) {
  int total = nrows * 320;
  for (int idx = blockIdx.x * blockDim.x + threadIdx.x; idx < total;
       idx += gridDim.x * blockDim.x) {
    int row = idx / 320, e = idx - row * 320;
    float v = 0.f;
    if (e < 300) v = emb[(size_t)tok[row] * 300 + e];
    out[idx] = f2bf(v);
  }
}

// ---------------------------------------------------------------------------
// Big x-projection GEMM:  Gx[R,384] = A[R,320] @ WxP + bias(col), bf16 out.
// Block = 256 (8 waves). unit = wave id over grid; mtile = unit>>1 (16 rows),
// nhalf = unit&1 (12 of 24 N-tiles). Single base pointer per operand with
// immediate offsets -> no per-tile pointer registers, no spills.
// ---------------------------------------------------------------------------
__global__ void xproj(const unsigned short* __restrict__ A,
                      const unsigned short* __restrict__ WP,
                      const float* __restrict__ gate_b,
                      const float* __restrict__ cand_b,
                      unsigned short* __restrict__ Gx) {
  int lane = threadIdx.x & 31;
  int wave = threadIdx.x >> 5;
  int unit = blockIdx.x * 8 + wave;
  int mtile = unit >> 1, nhalf = unit & 1;
  int m = lane & 15, half = lane >> 4;
  // A base: row (mtile*16+m), + 8*half elements (lane-half K swizzle folded in)
  const char* aptr = (const char*)(A + (size_t)(mtile * 16 + m) * 320 + 8 * half);
  // W base: this wave's N-half, this lane's 32B slice
  const char* wptr = (const char*)WP + (size_t)nhalf * 12 * 1024 + (size_t)lane * 32;
  v8f acc[12];
  v8f z = {0.f, 0.f, 0.f, 0.f, 0.f, 0.f, 0.f, 0.f};
#pragma unroll
  for (int j = 0; j < 12; j++) acc[j] = z;
  for (int kt = 0; kt < 10; kt++) {
    Frag16 af;
#pragma unroll
    for (int p = 0; p < 8; p++) af.u[p] = *(const unsigned int*)(aptr + ka_byte(p));
#pragma unroll
    for (int j = 0; j < 12; j++) {
      v16bf b = *(const v16bf*)(wptr + j * 1024);
      acc[j] = wmma_bf16(af.v, b, acc[j]);
    }
    aptr += 64;        // 32 bf16 per K-step
    wptr += 24 * 1024; // 24 N-tiles per K-step
  }
#pragma unroll
  for (int j = 0; j < 12; j++) {
    int n = (nhalf * 12 + j) * 16 + m;
    float bias = (n < 256) ? gate_b[n] : cand_b[n - 256];
#pragma unroll
    for (int r = 0; r < 8; r++) {
      unsigned row = mtile * 16 + r + 8 * half;
      Gx[(size_t)row * 384u + n] = f2bf(acc[j][r] + bias);
    }
  }
}

// ---------------------------------------------------------------------------
// Recurrent GRU over one (sequence, direction). One block = 16 batch rows,
// whole sequence. 8 waves; per step phase1 computes sigmoid gates (16 N-tiles),
// phase2 candidate tanh + state update (8 N-tiles). Wh (96 KB) is staged into
// LDS once; h double-buffered in LDS. reverse==1 reads/writes through the
// reverse_seq permutation. Next step's Gx tiles are prefetched.
// ---------------------------------------------------------------------------
__global__ void gru_seq(const unsigned short* __restrict__ Gx,
                        const unsigned short* __restrict__ WhP,
                        const int* __restrict__ lengths,
                        unsigned short* __restrict__ states,
                        int S, int reverse) {
  __shared__ unsigned short wh_sh[4 * 24 * 512];   // 96 KB packed Wh tiles
  __shared__ unsigned short h_sh[2][16][128];
  __shared__ unsigned short r_sh[16][128];
  __shared__ float u_sh[16][128];
  __shared__ int len_sh[16];
  int tid = threadIdx.x;
  int wave = tid >> 5, lane = tid & 31;
  int m = lane & 15, half = lane >> 4;
  int b0 = blockIdx.x * 16;
  if (tid < 16) len_sh[tid] = lengths[b0 + tid];
  for (int i = tid; i < 2048; i += 256) ((unsigned short*)h_sh[0])[i] = 0;
  for (int i = tid; i < 6144; i += 256) ((uint4*)wh_sh)[i] = ((const uint4*)WhP)[i];
  __syncthreads();
  int Lj[8];
#pragma unroll
  for (int j = 0; j < 8; j++) Lj[j] = len_sh[j + 8 * half];
  int ntA = 2 * wave, ntc = 16 + wave;
  int nA = ntA * 16 + m, nB = nA + 16;
  int nc = wave * 16 + m, nf = ntc * 16 + m;
  // LDS weight bases for this wave/lane (tile offsets are immediates below)
  const char* whA = (const char*)wh_sh + ntA * 1024 + lane * 32;
  const char* whC = (const char*)wh_sh + ntc * 1024 + lane * 32;
  const char* hrow = (const char*)&h_sh[0][m][0] + 16 * half;  // + cur*4096

  for (int t = 0; t < S; t++) {
    int cur = t & 1, nxt = cur ^ 1;
    unsigned rowoff[8];
#pragma unroll
    for (int j = 0; j < 8; j++) {
      int tr = t;
      if (reverse) tr = (t < Lj[j]) ? (Lj[j] - 1 - t) : t;
      rowoff[j] = ((unsigned)(b0 + j + 8 * half) * S + tr) * 384u;
      int t2 = (t + 1 < S) ? (t + 1) : t;   // prefetch next step's Gx row
      int tr2 = t2;
      if (reverse) tr2 = (t2 < Lj[j]) ? (Lj[j] - 1 - t2) : t2;
      __builtin_prefetch(Gx + ((unsigned)(b0 + j + 8 * half) * S + tr2) * 384u + nA, 0, 1);
    }
    // ---- phase 1: ru = sigmoid(Gx_gate + h @ Wh_gate) ----
    v8f acc0, acc1;
#pragma unroll
    for (int j = 0; j < 8; j++) {
      acc0[j] = bf2f(Gx[rowoff[j] + nA]);
      acc1[j] = bf2f(Gx[rowoff[j] + nB]);
    }
#pragma unroll
    for (int kt = 0; kt < 4; kt++) {
      Frag16 af;
#pragma unroll
      for (int p = 0; p < 8; p++)
        af.u[p] = *(const unsigned int*)(hrow + cur * 4096 + kt * 64 + ka_byte(p));
      v16bf bA = *(const v16bf*)(whA + kt * 24576);
      v16bf bB = *(const v16bf*)(whA + kt * 24576 + 1024);
      acc0 = wmma_bf16(af.v, bA, acc0);
      acc1 = wmma_bf16(af.v, bB, acc1);
    }
#pragma unroll
    for (int j = 0; j < 8; j++) {
      float g0 = 1.f / (1.f + __expf(-acc0[j]));
      float g1 = 1.f / (1.f + __expf(-acc1[j]));
      int row = j + 8 * half;
      if (wave < 4) {                     // wave-uniform branch: cols < 128 -> r
        r_sh[row][nA] = f2bf(g0);
        r_sh[row][nB] = f2bf(g1);
      } else {                            // cols 128..255 -> u
        u_sh[row][nA - 128] = g0;
        u_sh[row][nB - 128] = g1;
      }
    }
    __syncthreads();
    // ---- phase 2: c = tanh(Gx_cand + (r*h) @ Wh_cand); hn = u*h + (1-u)*c ----
    v8f acc;
#pragma unroll
    for (int j = 0; j < 8; j++) acc[j] = bf2f(Gx[rowoff[j] + nf]);
#pragma unroll
    for (int kt = 0; kt < 4; kt++) {
      Frag16 af;
#pragma unroll
      for (int p = 0; p < 8; p++) {
        int k = kt * 32 + (ka_byte(p) >> 1) + 8 * half;
        float rh0 = bf2f(r_sh[m][k])     * bf2f(h_sh[cur][m][k]);
        float rh1 = bf2f(r_sh[m][k + 1]) * bf2f(h_sh[cur][m][k + 1]);
        af.s[2 * p]     = f2bf(rh0);
        af.s[2 * p + 1] = f2bf(rh1);
      }
      v16bf bC = *(const v16bf*)(whC + kt * 24576);
      acc = wmma_bf16(af.v, bC, acc);
    }
#pragma unroll
    for (int j = 0; j < 8; j++) {
      int row = j + 8 * half;
      float c = tanhf(acc[j]);
      float hf = bf2f(h_sh[cur][row][nc]);
      float u = u_sh[row][nc];
      float hn = u * hf + (1.f - u) * c;
      bool act = (t < Lj[j]);
      float hnew = act ? hn : hf;
      float ov = act ? hn : 0.f;
      h_sh[nxt][row][nc] = f2bf(hnew);
      int sw = t;
      if (reverse && act) sw = Lj[j] - 1 - t;
      states[((unsigned)(b0 + row) * S + sw) * 128u + nc] = f2bf(ov);
    }
    __syncthreads();
  }
}

// ---------------------------------------------------------------------------
// Attention scores: score[row] = att_v . tanh(states[row,:] @ att_w + att_b)
// WMMA [16 rows,128] @ [128,128], then per-row cross-lane reduction.
// ---------------------------------------------------------------------------
__global__ void att_scores(const unsigned short* __restrict__ states,
                           const unsigned short* __restrict__ WP,
                           const float* __restrict__ att_v,
                           const float* __restrict__ att_b,
                           float* __restrict__ scores) {
  int lane = threadIdx.x & 31;
  int wave = threadIdx.x >> 5;
  int m = lane & 15, half = lane >> 4;
  int mtile = blockIdx.x * 8 + wave;
  const char* aptr = (const char*)(states + (size_t)(mtile * 16 + m) * 128 + 8 * half);
  const char* wptr = (const char*)WP + (size_t)lane * 32;
  v8f acc[8];
  v8f z = {0.f, 0.f, 0.f, 0.f, 0.f, 0.f, 0.f, 0.f};
#pragma unroll
  for (int nt = 0; nt < 8; nt++) acc[nt] = z;
#pragma unroll
  for (int kt = 0; kt < 4; kt++) {
    Frag16 af;
#pragma unroll
    for (int p = 0; p < 8; p++)
      af.u[p] = *(const unsigned int*)(aptr + kt * 64 + ka_byte(p));
#pragma unroll
    for (int nt = 0; nt < 8; nt++) {
      v16bf b = *(const v16bf*)(wptr + (kt * 8 + nt) * 1024);
      acc[nt] = wmma_bf16(af.v, b, acc[nt]);
    }
  }
  float rowsum[8];
#pragma unroll
  for (int j = 0; j < 8; j++) rowsum[j] = 0.f;
#pragma unroll
  for (int nt = 0; nt < 8; nt++) {
    int n = nt * 16 + m;
    float vb = att_b[n], vv = att_v[n];
#pragma unroll
    for (int j = 0; j < 8; j++) rowsum[j] += tanhf(acc[nt][j] + vb) * vv;
  }
#pragma unroll
  for (int j = 0; j < 8; j++) {
    float s = rowsum[j];
    s += __shfl_xor(s, 1, 32);
    s += __shfl_xor(s, 2, 32);
    s += __shfl_xor(s, 4, 32);
    s += __shfl_xor(s, 8, 32);
    rowsum[j] = s;
  }
  if (m == 0) {
#pragma unroll
    for (int j = 0; j < 8; j++)
      scores[(size_t)mtile * 16 + j + 8 * half] = rowsum[j];
  }
}

// ---------------------------------------------------------------------------
// Softmax over time (all S positions, matching reference) + weighted sum.
// One block per batch row; 128 threads = 128 channels.
// ---------------------------------------------------------------------------
__global__ void att_apply(const float* __restrict__ scores,
                          const unsigned short* __restrict__ states,
                          float* __restrict__ feats, int S, int featOff) {
  int b = blockIdx.x, tid = threadIdx.x;
  __shared__ float red[128];
  const float* sc = scores + (size_t)b * S;
  float mx = -3.0e38f;
  for (int s = tid; s < S; s += 128) mx = fmaxf(mx, sc[s]);
  red[tid] = mx;
  __syncthreads();
  for (int o = 64; o > 0; o >>= 1) {
    if (tid < o) red[tid] = fmaxf(red[tid], red[tid + o]);
    __syncthreads();
  }
  mx = red[0];
  __syncthreads();
  float se = 0.f;
  for (int s = tid; s < S; s += 128) se += __expf(sc[s] - mx);
  red[tid] = se;
  __syncthreads();
  for (int o = 64; o > 0; o >>= 1) {
    if (tid < o) red[tid] += red[tid + o];
    __syncthreads();
  }
  float inv = 1.f / red[0];
  const unsigned short* st = states + (size_t)b * S * 128 + tid;
  float accv = 0.f;
  for (int s = 0; s < S; s++) accv += __expf(sc[s] - mx) * bf2f(st[(size_t)s * 128]);
  feats[(size_t)b * 512 + featOff + tid] = accv * inv;
}

// ---------------------------------------------------------------------------
// Head: logit = relu(feats @ hid_w + hid_b) @ out_w + out_b  (tiny, VALU)
// ---------------------------------------------------------------------------
__global__ void head_kernel(const float* __restrict__ feats,
                            const float* __restrict__ hid_w,
                            const float* __restrict__ hid_b,
                            const float* __restrict__ out_w,
                            const float* __restrict__ out_b,
                            float* __restrict__ out) {
  int b = blockIdx.x, h = threadIdx.x;
  __shared__ float red[128];
  const float* f = feats + (size_t)b * 512;
  float a = hid_b[h];
  for (int k = 0; k < 512; k++) a = fmaf(f[k], hid_w[k * 128 + h], a);
  a = fmaxf(a, 0.f);
  red[h] = a * out_w[h];
  __syncthreads();
  for (int o = 64; o > 0; o >>= 1) {
    if (h < o) red[h] += red[h + o];
    __syncthreads();
  }
  if (h == 0) out[b] = red[0] + out_b[0];
}

// ---------------------------------------------------------------------------
extern "C" void kernel_launch(void* const* d_in, const int* in_sizes, int n_in,
                              void* d_out, int out_size, void* d_ws, size_t ws_size,
                              hipStream_t stream) {
  (void)in_sizes; (void)n_in; (void)out_size; (void)ws_size;
  const int B = 128, SC = 512, SE = 64;
  const int RC = B * SC, RE = B * SE;

  const int*   ctx     = (const int*)d_in[0];
  const int*   endt    = (const int*)d_in[1];
  const int*   ctx_len = (const int*)d_in[2];
  const int*   end_len = (const int*)d_in[3];
  const float* emb     = (const float*)d_in[4];
  const float* fw_gw   = (const float*)d_in[5];
  const float* fw_gb   = (const float*)d_in[6];
  const float* fw_cw   = (const float*)d_in[7];
  const float* fw_cb   = (const float*)d_in[8];
  const float* bw_gw   = (const float*)d_in[9];
  const float* bw_gb   = (const float*)d_in[10];
  const float* bw_cw   = (const float*)d_in[11];
  const float* bw_cb   = (const float*)d_in[12];
  const float* att_v   = (const float*)d_in[13];
  const float* att_w   = (const float*)d_in[14];
  const float* att_b   = (const float*)d_in[15];
  const float* hid_w   = (const float*)d_in[16];
  const float* hid_b   = (const float*)d_in[17];
  const float* out_w   = (const float*)d_in[18];
  const float* out_b   = (const float*)d_in[19];
  float* out = (float*)d_out;

  char* base = (char*)d_ws;
  size_t off = 0;
  auto alloc = [&](size_t bytes) -> char* {
    char* p = base + off;
    off = (off + bytes + 1023) & ~(size_t)1023;
    return p;
  };
  unsigned short* embC   = (unsigned short*)alloc((size_t)RC * 320 * 2);
  unsigned short* embE   = (unsigned short*)alloc((size_t)RE * 320 * 2);
  unsigned short* WxP_fw = (unsigned short*)alloc((size_t)10 * 24 * 512 * 2);
  unsigned short* WxP_bw = (unsigned short*)alloc((size_t)10 * 24 * 512 * 2);
  unsigned short* WhP_fw = (unsigned short*)alloc((size_t)4 * 24 * 512 * 2);
  unsigned short* WhP_bw = (unsigned short*)alloc((size_t)4 * 24 * 512 * 2);
  unsigned short* attWP  = (unsigned short*)alloc((size_t)4 * 8 * 512 * 2);
  unsigned short* GxC_fw = (unsigned short*)alloc((size_t)RC * 384 * 2);
  unsigned short* GxC_bw = (unsigned short*)alloc((size_t)RC * 384 * 2);
  unsigned short* GxE_fw = (unsigned short*)alloc((size_t)RE * 384 * 2);
  unsigned short* GxE_bw = (unsigned short*)alloc((size_t)RE * 384 * 2);
  unsigned short* stC_fw = (unsigned short*)alloc((size_t)RC * 128 * 2);
  unsigned short* stC_bw = (unsigned short*)alloc((size_t)RC * 128 * 2);
  unsigned short* stE_fw = (unsigned short*)alloc((size_t)RE * 128 * 2);
  unsigned short* stE_bw = (unsigned short*)alloc((size_t)RE * 128 * 2);
  float* scC_fw = (float*)alloc((size_t)RC * 4);
  float* scC_bw = (float*)alloc((size_t)RC * 4);
  float* scE_fw = (float*)alloc((size_t)RE * 4);
  float* scE_bw = (float*)alloc((size_t)RE * 4);
  float* feats  = (float*)alloc((size_t)B * 512 * 4);

  // 1) pack weights into WMMA B-fragment tile layout (bf16)
  pack_tiles<<<10 * 24, 512, 0, stream>>>(fw_gw, fw_cw, 256, 128, 24, 0, 300, WxP_fw);
  pack_tiles<<<4 * 24, 512, 0, stream>>>(fw_gw, fw_cw, 256, 128, 24, 300, 128, WhP_fw);
  pack_tiles<<<10 * 24, 512, 0, stream>>>(bw_gw, bw_cw, 256, 128, 24, 0, 300, WxP_bw);
  pack_tiles<<<4 * 24, 512, 0, stream>>>(bw_gw, bw_cw, 256, 128, 24, 300, 128, WhP_bw);
  pack_tiles<<<4 * 8, 512, 0, stream>>>(att_w, att_w, 128, 128, 8, 0, 128, attWP);

  // 2) embedding gather + bf16 cast (K padded to 320)
  embed_cast<<<2048, 256, 0, stream>>>(ctx, emb, embC, RC);
  embed_cast<<<2048, 256, 0, stream>>>(endt, emb, embE, RE);

  // 3) big parallel x-projection GEMMs (bias folded)
  xproj<<<RC / 64, 256, 0, stream>>>(embC, WxP_fw, fw_gb, fw_cb, GxC_fw);
  xproj<<<RC / 64, 256, 0, stream>>>(embC, WxP_bw, bw_gb, bw_cb, GxC_bw);
  xproj<<<RE / 64, 256, 0, stream>>>(embE, WxP_fw, fw_gb, fw_cb, GxE_fw);
  xproj<<<RE / 64, 256, 0, stream>>>(embE, WxP_bw, bw_gb, bw_cb, GxE_bw);

  // 4) sequential GRU recurrences (8 batch-tiles each, 4 independent launches)
  gru_seq<<<8, 256, 0, stream>>>(GxC_fw, WhP_fw, ctx_len, stC_fw, SC, 0);
  gru_seq<<<8, 256, 0, stream>>>(GxC_bw, WhP_bw, ctx_len, stC_bw, SC, 1);
  gru_seq<<<8, 256, 0, stream>>>(GxE_fw, WhP_fw, end_len, stE_fw, SE, 0);
  gru_seq<<<8, 256, 0, stream>>>(GxE_bw, WhP_bw, end_len, stE_bw, SE, 1);

  // 5) attention scores (WMMA) + softmax/weighted-sum into feature slices
  att_scores<<<RC / 128, 256, 0, stream>>>(stC_fw, attWP, att_v, att_b, scC_fw);
  att_scores<<<RC / 128, 256, 0, stream>>>(stC_bw, attWP, att_v, att_b, scC_bw);
  att_scores<<<RE / 128, 256, 0, stream>>>(stE_fw, attWP, att_v, att_b, scE_fw);
  att_scores<<<RE / 128, 256, 0, stream>>>(stE_bw, attWP, att_v, att_b, scE_bw);

  att_apply<<<B, 128, 0, stream>>>(scC_fw, stC_fw, feats, SC, 0);
  att_apply<<<B, 128, 0, stream>>>(scC_bw, stC_bw, feats, SC, 128);
  att_apply<<<B, 128, 0, stream>>>(scE_fw, stE_fw, feats, SE, 256);
  att_apply<<<B, 128, 0, stream>>>(scE_bw, stE_bw, feats, SE, 384);

  // 6) head
  head_kernel<<<B, 128, 0, stream>>>(feats, hid_w, hid_b, out_w, out_b, out);
}
